// Mamba2Block_20839181320237
// MI455X (gfx1250) — matrix operand order repaired and checked by reading, not compiled, not run on hardware
//
#include <hip/hip_runtime.h>
#include <cstdint>
#include <cstddef>

typedef __attribute__((ext_vector_type(4)))  float    v4f;
typedef __attribute__((ext_vector_type(8)))  float    v8f;
typedef __attribute__((ext_vector_type(16))) _Float16 v16h;

#define D_MODEL 1024
#define D_INNER 2048
#define D_STATE 16
#define DT_RANK 64
#define BATCH   2
#define SEQLEN  2048
#define NROWS   (BATCH * SEQLEN)          // 4096

// ---------------------------------------------------------------------------
// gfx1250 async global->LDS copies (ASYNCcnt path). Fallback = sync copy.
// Builtin signature (from hipcc diagnostic): arg0 = addrspace(1) src,
// arg1 = addrspace(3) dst, imm offset, imm cpol; pointees are int / int4.
// ---------------------------------------------------------------------------
#if defined(__gfx1250__) && \
    __has_builtin(__builtin_amdgcn_global_load_async_to_lds_b128) && \
    __has_builtin(__builtin_amdgcn_global_load_async_to_lds_b32)
#define USE_ASYNC_LDS 1
#else
#define USE_ASYNC_LDS 0
#endif

#if USE_ASYNC_LDS
typedef __attribute__((vector_size(16))) int v4i_gcc;
typedef __attribute__((address_space(1))) v4i_gcc* gv4i_p;   // global int4*
typedef __attribute__((address_space(3))) v4i_gcc* lv4i_p;   // lds    int4*
typedef __attribute__((address_space(1))) int*     gi32_p;   // global int*
typedef __attribute__((address_space(3))) int*     li32_p;   // lds    int*
#endif

__device__ __forceinline__ void async_copy_b128(const float* g, float* l) {
#if USE_ASYNC_LDS
  __builtin_amdgcn_global_load_async_to_lds_b128(
      (gv4i_p)(v4i_gcc*)(void*)const_cast<float*>(g),
      (lv4i_p)(v4i_gcc*)(void*)l, 0, 0);
#else
  *(v4f*)l = *(const v4f*)g;
#endif
}

__device__ __forceinline__ void async_copy_b32(const float* g, float* l) {
#if USE_ASYNC_LDS
  __builtin_amdgcn_global_load_async_to_lds_b32(
      (gi32_p)(int*)(void*)const_cast<float*>(g),
      (li32_p)(int*)(void*)l, 0, 0);
#else
  *l = *g;
#endif
}

__device__ __forceinline__ void async_wait_all() {
#if USE_ASYNC_LDS
#if __has_builtin(__builtin_amdgcn_s_wait_asynccnt)
  __builtin_amdgcn_s_wait_asynccnt(0);
#else
  asm volatile("s_wait_asynccnt 0x0" ::: "memory");
#endif
#endif
}

// ---------------------------------------------------------------------------
// LayerNorm over last dim (1024). One block per row.
// ---------------------------------------------------------------------------
__global__ __launch_bounds__(256) void ln_kernel(
    const float* __restrict__ x, const float* __restrict__ w,
    const float* __restrict__ b, float* __restrict__ out)
{
  __shared__ float red[16];
  const int row = blockIdx.x;
  const float* xr = x + (size_t)row * D_MODEL;
  float vals[4];
  float s = 0.f, ss = 0.f;
#pragma unroll
  for (int i = 0; i < 4; ++i) {
    float v = xr[threadIdx.x + 256 * i];
    vals[i] = v; s += v; ss += v * v;
  }
#pragma unroll
  for (int off = 16; off >= 1; off >>= 1) {
    s  += __shfl_xor(s,  off, 32);
    ss += __shfl_xor(ss, off, 32);
  }
  const int wave = threadIdx.x >> 5, lane = threadIdx.x & 31;
  if (lane == 0) { red[wave] = s; red[8 + wave] = ss; }
  __syncthreads();
  float ts = 0.f, tss = 0.f;
#pragma unroll
  for (int i = 0; i < 8; ++i) { ts += red[i]; tss += red[8 + i]; }
  const float mu  = ts * (1.f / D_MODEL);
  const float var = tss * (1.f / D_MODEL) - mu * mu;
  const float inv = rsqrtf(var + 1e-5f);
#pragma unroll
  for (int i = 0; i < 4; ++i) {
    const int c = threadIdx.x + 256 * i;
    out[(size_t)row * D_MODEL + c] = (vals[i] - mu) * inv * w[c] + b[c];
  }
}

// ---------------------------------------------------------------------------
// Generic TN GEMM on the matrix pipe: C[m,n] = sum_k A[m,k] * W[n,k]
// f32 in HBM, f16 fragments in LDS, f32 WMMA accumulation (16x16x32).
// Block tile 64(M) x 128(N), 8 waves, each wave a 32x32 tile (4 WMMAs/chunk).
// epi: 0 = plain, 1 = +bias then softplus, 2 = +residual
// ---------------------------------------------------------------------------
#define BM 64
#define BN 128
#define KT 32
#define LDSW (KT + 2)   // 34 halves = 68B row stride -> 17-bank stride (odd)

__device__ __forceinline__ float softplusf(float v) {
  return (v > 20.f) ? v : log1pf(__expf(v));
}

__global__ __launch_bounds__(256) void gemm_tn_wmma(
    const float* __restrict__ A, int lda,
    const float* __restrict__ W, int ldw,
    float* __restrict__ C, int ldc,
    int M, int N, int K,
    int epi,
    const float* __restrict__ bias,
    const float* __restrict__ res, int ldres)
{
  __shared__ _Float16 As[BM * LDSW];
  __shared__ _Float16 Ws[BN * LDSW];

  const int tid   = threadIdx.x;
  const int wave  = tid >> 5;
  const int lane  = tid & 31;
  const int nlo   = lane & 15;
  const int khalf = lane >> 4;

  const int bm = blockIdx.x * BM;
  const int bn = blockIdx.y * BN;

  const int wm = wave >> 2;           // 0..1 : 32-row strip
  const int wn = wave & 3;            // 0..3 : 32-col strip
  const int ra0 = 32 * wm + nlo;
  const int ra1 = ra0 + 16;
  const int rb0 = 32 * wn + nlo;
  const int rb1 = rb0 + 16;

  v8f acc00 = {0.f,0.f,0.f,0.f,0.f,0.f,0.f,0.f};
  v8f acc01 = acc00, acc10 = acc00, acc11 = acc00;

  const int r  = tid >> 2;        // 0..63
  const int c8 = (tid & 3) * 8;   // 0,8,16,24

  for (int k0 = 0; k0 < K; k0 += KT) {
    {   // A tile: 64 x 32
      const float* src = A + (size_t)(bm + r) * lda + k0 + c8;
      v4f x0 = *(const v4f*)(src);
      v4f x1 = *(const v4f*)(src + 4);
      _Float16* d = &As[r * LDSW + c8];
      d[0]=(_Float16)x0.x; d[1]=(_Float16)x0.y; d[2]=(_Float16)x0.z; d[3]=(_Float16)x0.w;
      d[4]=(_Float16)x1.x; d[5]=(_Float16)x1.y; d[6]=(_Float16)x1.z; d[7]=(_Float16)x1.w;
    }
#pragma unroll
    for (int half = 0; half < 2; ++half) {   // W tile: 128 x 32, two passes
      const int rr = r + 64 * half;
      v4f x0 = {0.f,0.f,0.f,0.f}, x1 = {0.f,0.f,0.f,0.f};
      if (bn + rr < N) {
        const float* src = W + (size_t)(bn + rr) * ldw + k0 + c8;
        x0 = *(const v4f*)(src);
        x1 = *(const v4f*)(src + 4);
      }
      _Float16* d = &Ws[rr * LDSW + c8];
      d[0]=(_Float16)x0.x; d[1]=(_Float16)x0.y; d[2]=(_Float16)x0.z; d[3]=(_Float16)x0.w;
      d[4]=(_Float16)x1.x; d[5]=(_Float16)x1.y; d[6]=(_Float16)x1.z; d[7]=(_Float16)x1.w;
    }
    __syncthreads();

    // Fragments per ISA 7.12.2 (16-bit A 16x32, B 32x16):
    v16h a0, a1, b0, b1;
#pragma unroll
    for (int j = 0; j < 16; ++j) {
      const int k = ((j >> 3) << 4) + (khalf << 3) + (j & 7);
      a0[j] = As[ra0 * LDSW + k];
      a1[j] = As[ra1 * LDSW + k];
    }
#pragma unroll
    for (int j = 0; j < 16; ++j) {
      const int k = ((j >> 3) << 4) + (khalf << 3) + (j & 7);
      b0[j] = Ws[rb0 * LDSW + k];
      b1[j] = Ws[rb1 * LDSW + k];
    }
    acc00 = __builtin_amdgcn_wmma_f32_16x16x32_f16(false, a0, false, b0,
                                                   (short)0, acc00, false, false);
    acc01 = __builtin_amdgcn_wmma_f32_16x16x32_f16(false, a0, false, b1,
                                                   (short)0, acc01, false, false);
    acc10 = __builtin_amdgcn_wmma_f32_16x16x32_f16(false, a1, false, b0,
                                                   (short)0, acc10, false, false);
    acc11 = __builtin_amdgcn_wmma_f32_16x16x32_f16(false, a1, false, b1,
                                                   (short)0, acc11, false, false);
    __syncthreads();
  }

  // C/D layout: VGPR i -> m = i + 8*(lane/16), n = lane%16
  const int n0 = bn + 32 * wn + nlo;
  const int n1 = n0 + 16;
  const bool ok0 = n0 < N, ok1 = n1 < N;
  const float bb0 = (epi == 1 && ok0) ? bias[n0] : 0.f;
  const float bb1 = (epi == 1 && ok1) ? bias[n1] : 0.f;
#pragma unroll
  for (int i = 0; i < 8; ++i) {
    const int m0 = bm + 32 * wm + i + (khalf << 3);
    const int m1 = m0 + 16;
    float v00 = acc00[i], v01 = acc01[i];
    float v10 = acc10[i], v11 = acc11[i];
    if (epi == 1) {
      v00 = softplusf(v00 + bb0); v01 = softplusf(v01 + bb1);
      v10 = softplusf(v10 + bb0); v11 = softplusf(v11 + bb1);
    } else if (epi == 2) {
      if (ok0) { v00 += res[(size_t)m0 * ldres + n0]; v10 += res[(size_t)m1 * ldres + n0]; }
      if (ok1) { v01 += res[(size_t)m0 * ldres + n1]; v11 += res[(size_t)m1 * ldres + n1]; }
    }
    if (ok0) { C[(size_t)m0 * ldc + n0] = v00; C[(size_t)m1 * ldc + n0] = v10; }
    if (ok1) { C[(size_t)m0 * ldc + n1] = v01; C[(size_t)m1 * ldc + n1] = v11; }
  }
}

// ---------------------------------------------------------------------------
// Depthwise causal conv (K=4) + bias + SiLU.  x lives in xz[:, 0:2048].
// ---------------------------------------------------------------------------
__global__ __launch_bounds__(256) void conv_silu_kernel(
    const float* __restrict__ xz, const float* __restrict__ cw,
    const float* __restrict__ cb, float* __restrict__ xc)
{
  const int idx = blockIdx.x * 256 + threadIdx.x;     // over B*L*ED
  const int e = idx & (D_INNER - 1);
  const int t = (idx >> 11) & (SEQLEN - 1);
  const int b = idx >> 22;
  float acc = cb[e];
#pragma unroll
  for (int j = 0; j < 4; ++j) {
    const int tt = t - 3 + j;
    if (tt >= 0)
      acc += cw[e * 4 + j] * xz[((size_t)(b * SEQLEN + tt)) * (2 * D_INNER) + e];
  }
  const float sig = 1.f / (1.f + __expf(-acc));
  xc[(size_t)idx] = acc * sig;
}

// ---------------------------------------------------------------------------
// Selective scan. 16 lanes per (b, e): one state per lane. Chunks of 32
// timesteps staged to LDS via gfx1250 async global->LDS copies (ASYNCcnt).
// Per step: h = exp(dt*A_n)*h + dt*x*B_n ; y = sum_n h*C_n (shfl_xor reduce).
// Fuses y += x*D and y *= silu(z).
// ---------------------------------------------------------------------------
#define TCH 32
__global__ __launch_bounds__(256) void scan_kernel(
    const float* __restrict__ dtb,    // (B*L, 4096) stride-4096, cols 0..2047
    const float* __restrict__ xc,     // (B*L, 2048)
    const float* __restrict__ xdbl,   // (B*L, 96): [64..79]=B, [80..95]=C
    const float* __restrict__ xz,     // z at cols 2048..4095
    const float* __restrict__ A_log,
    const float* __restrict__ Dp,
    float* __restrict__ y)
{
  __shared__ float sBC[TCH][32];
  __shared__ float sDT[TCH][16];
  __shared__ float sX [TCH][16];

  const int tid  = threadIdx.x;
  const int p    = tid >> 4;                 // channel slot in block, 0..15
  const int n    = tid & 15;                 // state index
  const int pair = blockIdx.x * 16 + p;      // (b,e) flat
  const int b    = pair >> 11;
  const int e    = pair & (D_INNER - 1);
  const int e0   = (blockIdx.x * 16) & (D_INNER - 1);

  const float An = -__expf(A_log[e * D_STATE + n]);
  const float De = Dp[e];
  const size_t rowbase = (size_t)b * SEQLEN;

  // staging indices (constant across chunks)
  const int tlBC = tid >> 3;
  const int cBC  = (tid & 7) * 4;
  const int ppS  = tid & 15;
  const int tlS  = tid >> 4;

  float h = 0.f;
  for (int t0 = 0; t0 < SEQLEN; t0 += TCH) {
    // stage B,C: 32 t x 32 vals, one b128 async copy per thread
    async_copy_b128(xdbl + (rowbase + t0 + tlBC) * 96 + DT_RANK + cBC,
                    &sBC[tlBC][cBC]);
    // stage dt, x: 32 t x 16 channels, four b32 async copies per thread
    async_copy_b32(dtb + (rowbase + t0 + tlS     ) * (2*D_INNER) + e0 + ppS, &sDT[tlS     ][ppS]);
    async_copy_b32(dtb + (rowbase + t0 + tlS + 16) * (2*D_INNER) + e0 + ppS, &sDT[tlS + 16][ppS]);
    async_copy_b32(xc  + (rowbase + t0 + tlS     ) * D_INNER + e0 + ppS,     &sX [tlS     ][ppS]);
    async_copy_b32(xc  + (rowbase + t0 + tlS + 16) * D_INNER + e0 + ppS,     &sX [tlS + 16][ppS]);
    async_wait_all();
    __syncthreads();

#pragma unroll 4
    for (int tl = 0; tl < TCH; ++tl) {
      const float dt = sDT[tl][p];
      const float xv = sX [tl][p];
      const float dA = __expf(dt * An);
      h = dA * h + (dt * xv) * sBC[tl][n];
      float yv = h * sBC[tl][16 + n];
      yv += __shfl_xor(yv, 1, 32);
      yv += __shfl_xor(yv, 2, 32);
      yv += __shfl_xor(yv, 4, 32);
      yv += __shfl_xor(yv, 8, 32);
      if (n == 0) {
        const int t = t0 + tl;
        const float z   = xz[(rowbase + t) * (2*D_INNER) + D_INNER + e];
        const float sig = 1.f / (1.f + __expf(-z));
        y[(rowbase + t) * D_INNER + e] = (yv + xv * De) * (z * sig);
      }
    }
    __syncthreads();
  }
}

// ---------------------------------------------------------------------------
extern "C" void kernel_launch(void* const* d_in, const int* in_sizes, int n_in,
                              void* d_out, int out_size, void* d_ws, size_t ws_size,
                              hipStream_t stream) {
  const float* hidden   = (const float*)d_in[0];
  const float* ln_w     = (const float*)d_in[1];
  const float* ln_b     = (const float*)d_in[2];
  const float* in_proj  = (const float*)d_in[3];   // (4096, 1024)
  const float* conv_w   = (const float*)d_in[4];   // (2048, 4)
  const float* conv_b   = (const float*)d_in[5];
  const float* x_proj   = (const float*)d_in[6];   // (96, 2048)
  const float* dt_proj  = (const float*)d_in[7];   // (2048, 64)
  const float* dt_bias  = (const float*)d_in[8];
  const float* A_log    = (const float*)d_in[9];
  const float* D_param  = (const float*)d_in[10];
  const float* out_proj = (const float*)d_in[11];  // (1024, 2048)
  float* out = (float*)d_out;

  // workspace overlays (floats):
  float* buf_xz = (float*)d_ws;                         // 4096 x 4096 (x|z; x-half reused for dt)
  float* buf_h  = buf_xz + (size_t)NROWS * 2 * D_INNER; // 4096 x 1024 (h_norm; reused for x_dbl)
  float* buf_xc = buf_h  + (size_t)NROWS * D_MODEL;     // 4096 x 2048
  float* buf_y  = buf_xc + (size_t)NROWS * D_INNER;     // 4096 x 2048

  // 1. LayerNorm
  ln_kernel<<<NROWS, 256, 0, stream>>>(hidden, ln_w, ln_b, buf_h);

  // 2. in_proj: (4096x1024) x (4096x1024)^T -> xz (4096x4096)
  gemm_tn_wmma<<<dim3(NROWS / BM, (2 * D_INNER) / BN), 256, 0, stream>>>(
      buf_h, D_MODEL, in_proj, D_MODEL, buf_xz, 2 * D_INNER,
      NROWS, 2 * D_INNER, D_MODEL, 0, nullptr, nullptr, 0);

  // 3. depthwise causal conv + SiLU -> buf_xc
  conv_silu_kernel<<<(NROWS * D_INNER) / 256, 256, 0, stream>>>(
      buf_xz, conv_w, conv_b, buf_xc);

  // 4. x_proj: (4096x2048) x (96x2048)^T -> x_dbl (4096x96) into buf_h
  gemm_tn_wmma<<<dim3(NROWS / BM, (96 + BN - 1) / BN), 256, 0, stream>>>(
      buf_xc, D_INNER, x_proj, D_INNER, buf_h, 96,
      NROWS, 96, D_INNER, 0, nullptr, nullptr, 0);

  // 5. dt_proj + bias + softplus -> dt into x-half of buf_xz (ldc = 4096)
  gemm_tn_wmma<<<dim3(NROWS / BM, D_INNER / BN), 256, 0, stream>>>(
      buf_h, 96, dt_proj, DT_RANK, buf_xz, 2 * D_INNER,
      NROWS, D_INNER, DT_RANK, 1, dt_bias, nullptr, 0);

  // 6. selective scan (+ D skip, + silu(z) gate) -> buf_y
  scan_kernel<<<(BATCH * D_INNER) / 16, 256, 0, stream>>>(
      buf_xz, buf_xc, buf_h, buf_xz, A_log, D_param, buf_y);

  // 7. out_proj + residual -> out
  gemm_tn_wmma<<<dim3(NROWS / BM, D_MODEL / BN), 256, 0, stream>>>(
      buf_y, D_INNER, out_proj, D_INNER, out, D_MODEL,
      NROWS, D_MODEL, D_INNER, 2, nullptr, hidden, D_MODEL);
}
